// CAGNNLayer_9749575762774
// MI455X (gfx1250) — compile-verified, matrix operand order checked by reading, not run on hardware
//
#include <hip/hip_runtime.h>

typedef __attribute__((ext_vector_type(16))) _Float16 v16h;
typedef __attribute__((ext_vector_type(8)))  _Float16 v8h;
typedef __attribute__((ext_vector_type(4)))  _Float16 v4h;
typedef __attribute__((ext_vector_type(8)))  float    v8f;
typedef __attribute__((ext_vector_type(4)))  float    v4f;

#define LN_EPS 1e-5f

// ---------------------------------------------------------------------------
// Pack a row-major f32 weight [K, 128] into f16 WMMA B-fragment order.
// Fragment = 32(K) x 16(N) tile, 512 halfs, laid out so lane L's 16 halfs are
// contiguous:  halfs 0..7  -> K = k0 + h      + (L>=16 ? 8 : 0),
//              halfs 8..15 -> K = k0 + (h-8) + 16 + (L>=16 ? 8 : 0),
//              column n = n0 + (L & 15).       (ISA 7.12.2, 16-bit layouts)
// Fragment index = kt * 8 + nt   (N fixed at 128 -> 8 n-tiles).
// ---------------------------------------------------------------------------
__global__ void pack_w_kernel(const float* __restrict__ W,
                              _Float16* __restrict__ P, int K) {
  int tid = blockIdx.x * 256 + threadIdx.x;
  int total = K * 128;
  if (tid >= total) return;
  int frag = tid >> 9;            // 512 halfs per fragment
  int lane = (tid >> 4) & 31;
  int h    = tid & 15;
  int kt = frag >> 3;
  int nt = frag & 7;
  int k = kt * 32 + (h & 7) + ((h >> 3) << 4) + ((lane >> 4) << 3);
  int n = nt * 16 + (lane & 15);
  P[tid] = (_Float16)W[k * 128 + n];
}

// ---------------------------------------------------------------------------
// Fused stage:  out = LayerNorm(relu(relu([self | mean-gather] @ Wagg + bAgg)
//                                      @ Wcom + bCom))
// One workgroup = 64 rows, 256 threads (8 wave32s). Wave w owns output
// columns [16w, 16w+16) and iterates the 4 m-tiles.
// ---------------------------------------------------------------------------
__global__ __launch_bounds__(256)
void fused_stage_kernel(const float* __restrict__ selfFeat,   // [rows,128]
                        const float* __restrict__ gatherSrc,  // [gRows,128]
                        const int*   __restrict__ nbrIdx,     // [rows,16]
                        const _Float16* __restrict__ Wagg,    // [8][8][512] f16 frags
                        const float* __restrict__ bAgg,       // [128]
                        const _Float16* __restrict__ Wcom,    // [4][8][512] f16 frags
                        const float* __restrict__ bCom,       // [128]
                        const float* __restrict__ gamma,      // [128]
                        const float* __restrict__ beta,       // [128]
                        float* __restrict__ outF,             // [rows,128]
                        int rows, int gRows) {
  __shared__ __align__(16) _Float16 X[64 * 264];  // concat input, f16 (stride 264)
  __shared__ __align__(16) _Float16 Y[64 * 136];  // hidden, f16 (stride 136)
  float* Z = (float*)X;                           // f32 overlay, stride 132 (X dead)

  const int tid  = threadIdx.x;
  const int wave = tid >> 5;
  const int lane = tid & 31;
  const int tileBase = blockIdx.x * 64;

  // ---- Phase 0: self features + mean of 16 gathered neighbor rows -> X ----
  {
    const int c = lane * 4;  // 4 floats per lane covers one 128-wide row
    for (int lr = wave * 8; lr < wave * 8 + 8; ++lr) {
      int gr = tileBase + lr;
      if (gr >= rows) gr = rows - 1;  // duplicate work, stores guarded later
      v4f s = *(const v4f*)(selfFeat + (long)gr * 128 + c);
      *(v4h*)(X + lr * 264 + c) = __builtin_convertvector(s, v4h);
      v4f acc = {0.f, 0.f, 0.f, 0.f};
      const int* ip = nbrIdx + (long)gr * 16;
#pragma unroll
      for (int j = 0; j < 16; ++j) {
        int idx = ip[j];
        if (idx < 0) idx = 0;
        if (idx >= gRows) idx = gRows - 1;
        acc += *(const v4f*)(gatherSrc + (long)idx * 128 + c);
      }
      acc = acc * 0.0625f;
      *(v4h*)(X + lr * 264 + 128 + c) = __builtin_convertvector(acc, v4h);
    }
  }
  __syncthreads();

  const int col  = lane & 15;
  const int hoff = (lane >> 4) << 3;  // 0 for lanes 0-15, 8 for lanes 16-31
  const int n0   = wave * 16;

  // ---- Phase 1: GEMM1  X[64x256] @ Wagg[256x128], relu -> Y (f16) ----
  v8f acc1[4] = {};
#pragma unroll
  for (int kt = 0; kt < 8; ++kt) {
    const int k0 = kt * 32;
    const _Float16* bp = Wagg + (((kt << 3) + wave) << 9) + (lane << 4);
    v8h blo = *(const v8h*)bp;
    v8h bhi = *(const v8h*)(bp + 8);
    v16h b = __builtin_shufflevector(blo, bhi, 0, 1, 2, 3, 4, 5, 6, 7,
                                     8, 9, 10, 11, 12, 13, 14, 15);
    // Hoist all independent A-fragment loads so the DS clause overlaps WMMA.
    v16h afrag[4];
#pragma unroll
    for (int m = 0; m < 4; ++m) {
      const _Float16* ap = X + (m * 16 + col) * 264 + k0 + hoff;
      v8h alo = *(const v8h*)ap;
      v8h ahi = *(const v8h*)(ap + 16);
      afrag[m] = __builtin_shufflevector(alo, ahi, 0, 1, 2, 3, 4, 5, 6, 7,
                                         8, 9, 10, 11, 12, 13, 14, 15);
    }
#pragma unroll
    for (int m = 0; m < 4; ++m)
      acc1[m] = __builtin_amdgcn_wmma_f32_16x16x32_f16(
          false, afrag[m], false, b, (short)0, acc1[m], false, false);
  }
  {
    const float bb = bAgg[n0 + col];
#pragma unroll
    for (int m = 0; m < 4; ++m)
#pragma unroll
      for (int v = 0; v < 8; ++v) {
        float x = acc1[m][v] + bb;
        x = fmaxf(x, 0.f);
        Y[(m * 16 + v + hoff) * 136 + n0 + col] = (_Float16)x;
      }
  }
  __syncthreads();

  // ---- Phase 2: GEMM2  Y[64x128] @ Wcom[128x128], relu -> Z (f32) ----
  v8f acc2[4] = {};
#pragma unroll
  for (int kt = 0; kt < 4; ++kt) {
    const int k0 = kt * 32;
    const _Float16* bp = Wcom + (((kt << 3) + wave) << 9) + (lane << 4);
    v8h blo = *(const v8h*)bp;
    v8h bhi = *(const v8h*)(bp + 8);
    v16h b = __builtin_shufflevector(blo, bhi, 0, 1, 2, 3, 4, 5, 6, 7,
                                     8, 9, 10, 11, 12, 13, 14, 15);
    v16h afrag[4];
#pragma unroll
    for (int m = 0; m < 4; ++m) {
      const _Float16* ap = Y + (m * 16 + col) * 136 + k0 + hoff;
      v8h alo = *(const v8h*)ap;
      v8h ahi = *(const v8h*)(ap + 16);
      afrag[m] = __builtin_shufflevector(alo, ahi, 0, 1, 2, 3, 4, 5, 6, 7,
                                         8, 9, 10, 11, 12, 13, 14, 15);
    }
#pragma unroll
    for (int m = 0; m < 4; ++m)
      acc2[m] = __builtin_amdgcn_wmma_f32_16x16x32_f16(
          false, afrag[m], false, b, (short)0, acc2[m], false, false);
  }
  {
    const float bb = bCom[n0 + col];
#pragma unroll
    for (int m = 0; m < 4; ++m)
#pragma unroll
      for (int v = 0; v < 8; ++v) {
        float x = acc2[m][v] + bb;
        x = fmaxf(x, 0.f);
        Z[(m * 16 + v + hoff) * 132 + n0 + col] = x;
      }
  }
  __syncthreads();

  // ---- Phase 3: LayerNorm per row, write out ----
  {
    const int c = lane * 4;
    v4f g4 = *(const v4f*)(gamma + c);
    v4f b4 = *(const v4f*)(beta + c);
    for (int lr = wave * 8; lr < wave * 8 + 8; ++lr) {
      int gr = tileBase + lr;
      v4f z = *(const v4f*)(Z + lr * 132 + c);
      float s = z.x + z.y + z.z + z.w;
#pragma unroll
      for (int off = 16; off > 0; off >>= 1) s += __shfl_xor(s, off, 32);
      float mu = s * (1.f / 128.f);
      v4f d = z - mu;
      float q = d.x * d.x + d.y * d.y + d.z * d.z + d.w * d.w;
#pragma unroll
      for (int off = 16; off > 0; off >>= 1) q += __shfl_xor(q, off, 32);
      float r = rsqrtf(q * (1.f / 128.f) + LN_EPS);
      if (gr < rows) {
        v4f o = d * r * g4 + b4;
        *(v4f*)(outF + (long)gr * 128 + c) = o;
      }
    }
  }
}

// ---------------------------------------------------------------------------
extern "C" void kernel_launch(void* const* d_in, const int* in_sizes, int n_in,
                              void* d_out, int out_size, void* d_ws, size_t ws_size,
                              hipStream_t stream) {
  const int*   node_nbr   = (const int*)d_in[0];
  const int*   edge_nbr   = (const int*)d_in[1];
  const float* node_feats = (const float*)d_in[2];
  const float* edge_feats = (const float*)d_in[3];
  const float* W_edge_agg = (const float*)d_in[4];
  const float* b_edge_agg = (const float*)d_in[5];
  const float* W_edge_com = (const float*)d_in[6];
  const float* b_edge_com = (const float*)d_in[7];
  const float* W_node_agg = (const float*)d_in[8];
  const float* b_node_agg = (const float*)d_in[9];
  const float* W_node_com = (const float*)d_in[10];
  const float* b_node_com = (const float*)d_in[11];
  const float* ln_gamma   = (const float*)d_in[12];
  const float* ln_beta    = (const float*)d_in[13];

  const int N = in_sizes[2] / 128;
  const int E = in_sizes[3] / 128;

  float* out_nodes = (float*)d_out;                 // [N,128]
  float* out_edges = out_nodes + (size_t)N * 128;   // [E,128]

  char* ws = (char*)d_ws;
  _Float16* Pea = (_Float16*)(ws);            // 256x128 f16 = 64 KiB
  _Float16* Pec = (_Float16*)(ws + 65536);    // 128x128 f16 = 32 KiB
  _Float16* Pna = (_Float16*)(ws + 98304);    // 64 KiB
  _Float16* Pnc = (_Float16*)(ws + 163840);   // 32 KiB

  pack_w_kernel<<<(256 * 128 + 255) / 256, 256, 0, stream>>>(W_edge_agg, Pea, 256);
  pack_w_kernel<<<(128 * 128 + 255) / 256, 256, 0, stream>>>(W_edge_com, Pec, 128);
  pack_w_kernel<<<(256 * 128 + 255) / 256, 256, 0, stream>>>(W_node_agg, Pna, 256);
  pack_w_kernel<<<(128 * 128 + 255) / 256, 256, 0, stream>>>(W_node_com, Pnc, 128);

  // Edge stage: gather from edge_feats itself, write new_edge_feats.
  fused_stage_kernel<<<(E + 63) / 64, 256, 0, stream>>>(
      edge_feats, edge_feats, edge_nbr, Pea, b_edge_agg, Pec, b_edge_com,
      ln_gamma, ln_beta, out_edges, E, E);

  // Node stage: gather from the just-produced new_edge_feats.
  fused_stage_kernel<<<(N + 63) / 64, 256, 0, stream>>>(
      node_feats, out_edges, node_nbr, Pna, b_node_agg, Pnc, b_node_com,
      ln_gamma, ln_beta, out_nodes, N, E);
}
